// HGT_6373731467682
// MI455X (gfx1250) — matrix operand order repaired and checked by reading, not compiled, not run on hardware
//
#include <hip/hip_runtime.h>
#include <hip/hip_bf16.h>
#include <math.h>

// ---------------------------------------------------------------------------
// HGT forward for MI455X (gfx1250, wave32).
//  - Dense GEMMs: v_wmma_f32_16x16x32_bf16, bf16 operands pre-converted once.
//    A tiles staged global->LDS with global_load_async_to_lds_b128 (ASYNCcnt),
//    fragments fetched as aligned 16B chunks (ds_load_b128, ISA 7.12.2 layout).
//  - Edge softmax/aggregation: fp32 + global atomics (memory-bound phase).
// ---------------------------------------------------------------------------

typedef __attribute__((ext_vector_type(16))) __bf16 v16bf;
typedef __attribute__((ext_vector_type(8)))  float  v8f;

struct __align__(16) B128 { unsigned u0, u1, u2, u3; };
struct __align__(16) BF8  { __bf16 h[8]; };
union  FragU { B128 b[2]; v16bf v; };

#define NF_N   40000
#define NH_N   10000
#define EDGES  400000
#define IN_DIM 128
#define HID    256
#define HEADS  8
#define DH     32
#define LAYERS 2
#define OUT_DIM 64

// GEMM tiling: 256 threads = 8 waves arranged 2(M) x 4(N).
// Each wave owns a 32x32 C tile => four 16x16x32 bf16 WMMAs per K-step.
#define BM 64
#define BN 128
#define BK 32
#define LDS_STRIDE 40           // padded row stride in bf16 (80B, 16B multiple)

enum { EPI_NONE = 0, EPI_RELU = 1, EPI_BLEND = 2 };

__device__ __forceinline__ float gelu_exact(float x) {
  return 0.5f * x * (1.0f + erff(x * 0.70710678118654752440f));
}

// ---------------------------------------------------------------------------
// Fused WMMA GEMM:  C[M,N] = epi( A[M,K](bf16) @ W[K,N](bf16) + bias )
//  - blockIdx.z batches heads via flat element offsets (relation transforms).
//  - C (f32) and Cbf (bf16 mirror) are each optional.
//  - EPI_BLEND: v = sg*v + (1-sg)*Xold  (C may alias Xold; 1:1 thread mapping)
// ---------------------------------------------------------------------------
__global__ __launch_bounds__(256) void wmma_gemm_kernel(
    const __bf16* __restrict__ A, const __bf16* __restrict__ W,
    const float* __restrict__ bias, const float* __restrict__ Xold,
    const float* __restrict__ skipPtr, int skipIdx,
    float* __restrict__ C, __bf16* __restrict__ Cbf,
    int M, int K, int N, int lda, int ldw, int ldc,
    int aOffZ, int wOffZ, int cOffZ, int epi)
{
  __shared__ __bf16 lA[BM * LDS_STRIDE];    // A tile, row-major [m][k]
  __shared__ __bf16 lBt[BN * LDS_STRIDE];   // B tile, transposed [n][k]

  const int z = blockIdx.z;
  A += (size_t)z * aOffZ;
  W += (size_t)z * wOffZ;
  const size_t zC = (size_t)z * cOffZ;

  const int tid   = threadIdx.x;
  const int lane  = tid & 31;
  const int wave  = tid >> 5;
  const int waveM = wave >> 2;   // 0..1
  const int waveN = wave & 3;    // 0..3
  const int hlf   = lane >> 4;   // lane half selects K sub-range (ISA 7.12.2)
  const int l16   = lane & 15;

  const int rowBase = blockIdx.y * BM;
  const int colBase = blockIdx.x * BN;

  const char* lAc = (const char*)lA;
  const char* lBc = (const char*)lBt;
  const unsigned ldsA_base = (unsigned)(size_t)(&lA[0]);

  // A staging: one 16B chunk per thread per K-step (64 rows x 4 chunks = 256)
  const int ar  = tid >> 2;
  const int ach = tid & 3;
  int agr = rowBase + ar; if (agr >= M) agr = M - 1;   // clamp (rows guarded at store)
  const unsigned ldsA_off = ldsA_base + (unsigned)(ar * (LDS_STRIDE * 2) + ach * 16);

  v8f acc[2][2];
#pragma unroll
  for (int a = 0; a < 2; ++a)
#pragma unroll
    for (int b = 0; b < 2; ++b)
#pragma unroll
      for (int i = 0; i < 8; ++i) acc[a][b][i] = 0.0f;

  for (int k0 = 0; k0 < K; k0 += BK) {
    // ---- A tile: async global -> LDS (tracked by ASYNCcnt) ----
    {
      unsigned long long ga =
          (unsigned long long)(size_t)(A + (size_t)agr * lda + k0 + ach * 8);
      asm volatile("global_load_async_to_lds_b128 %0, %1, off"
                   :: "v"(ldsA_off), "v"(ga) : "memory");
    }
    // ---- B tile: 16B-wide global reads, transposed scatter into LDS ----
#pragma unroll
    for (int i = 0; i < 2; ++i) {
      int cid = tid + i * 256;          // 0..511
      int r   = cid >> 4;               // k row 0..31
      int c8  = (cid & 15) << 3;        // col chunk (8 bf16)
      BF8 w8;
      if (colBase + c8 + 8 <= N) {
        w8 = *(const BF8*)(W + (size_t)(k0 + r) * ldw + colBase + c8);
      } else {
#pragma unroll
        for (int j = 0; j < 8; ++j) w8.h[j] = (__bf16)0.0f;
      }
#pragma unroll
      for (int j = 0; j < 8; ++j) lBt[(c8 + j) * LDS_STRIDE + r] = w8.h[j];
    }
    asm volatile("s_wait_asynccnt 0x0" ::: "memory");
    __syncthreads();

    // ---- fragments: contiguous aligned 16B chunks -> ds_load_b128 ----
    v16bf afrag[2], bfrag[2];
#pragma unroll
    for (int sm = 0; sm < 2; ++sm) {
      int row = waveM * 32 + sm * 16 + l16;
      const char* p = lAc + row * (LDS_STRIDE * 2) + hlf * 16;
      FragU f;
      f.b[0] = *(const B128*)(p);        // e0..7  : k = hlf*8 + 0..7
      f.b[1] = *(const B128*)(p + 32);   // e8..15 : k = 16 + hlf*8 + 0..7
      afrag[sm] = f.v;
    }
#pragma unroll
    for (int sn = 0; sn < 2; ++sn) {
      int row = waveN * 32 + sn * 16 + l16;
      const char* p = lBc + row * (LDS_STRIDE * 2) + hlf * 16;
      FragU f;
      f.b[0] = *(const B128*)(p);        // e0..7  : k = hlf*8 + 0..7
      f.b[1] = *(const B128*)(p + 32);   // e8..15 : k = 16 + hlf*8 + 0..7
      bfrag[sn] = f.v;
    }

#pragma unroll
    for (int sm = 0; sm < 2; ++sm)
#pragma unroll
      for (int sn = 0; sn < 2; ++sn)
        acc[sm][sn] = __builtin_amdgcn_wmma_f32_16x16x32_bf16(
            false, afrag[sm], false, bfrag[sn], (short)0, acc[sm][sn],
            false, false);
    __syncthreads();   // protect LDS before next-step staging
  }

  // ---- epilogue ----
  float sg = 1.0f, omsg = 0.0f;
  if (epi == EPI_BLEND) {
    float s = skipPtr[skipIdx];
    sg = 1.0f / (1.0f + __expf(-s));
    omsg = 1.0f - sg;
  }
#pragma unroll
  for (int sm = 0; sm < 2; ++sm) {
#pragma unroll
    for (int sn = 0; sn < 2; ++sn) {
      int colG = colBase + waveN * 32 + sn * 16 + l16;
#pragma unroll
      for (int i = 0; i < 8; ++i) {
        int rowG = rowBase + waveM * 32 + sm * 16 + hlf * 8 + i;
        if (rowG < M && colG < N) {
          float v = acc[sm][sn][i];
          if (bias) v += bias[colG];
          if (epi == EPI_RELU)       v = fmaxf(v, 0.0f);
          else if (epi == EPI_BLEND)
            v = sg * v + omsg * Xold[(size_t)rowG * ldc + colG];
          size_t idx = zC + (size_t)rowG * ldc + colG;
          if (C)   C[idx]   = v;
          if (Cbf) Cbf[idx] = (__bf16)v;
        }
      }
    }
  }
}

// ---------------------------------------------------------------------------
// Elementwise helpers
// ---------------------------------------------------------------------------
__global__ void fill_kernel(float* __restrict__ p, int n, float v) {
  int i = blockIdx.x * blockDim.x + threadIdx.x;
  if (i < n) p[i] = v;
}

// dst = bf16(act(src));  act: 0 = identity, 1 = exact GELU
__global__ void convert_bf16_kernel(const float* __restrict__ src,
                                    __bf16* __restrict__ dst, int n, int act) {
  int i = blockIdx.x * blockDim.x + threadIdx.x;
  if (i >= n) return;
  float v = src[i];
  if (act) v = gelu_exact(v);
  dst[i] = (__bf16)v;
}

// ---------------------------------------------------------------------------
// Edge-phase kernels (memory/atomic bound, fp32)
// ---------------------------------------------------------------------------
__device__ __forceinline__ void atomicMaxF32(float* addr, float val) {
  if (val >= 0.0f) atomicMax((int*)addr, __float_as_int(val));
  else             atomicMin((unsigned int*)addr, __float_as_uint(val));
}

__global__ void edge_score_kernel(const int* __restrict__ ei,
                                  const float* __restrict__ q,
                                  const float* __restrict__ kt,
                                  const float* __restrict__ prel,
                                  float* __restrict__ score,
                                  float* __restrict__ mx)
{
  int t = blockIdx.x * blockDim.x + threadIdx.x;
  if (t >= EDGES * HEADS) return;
  int e = t >> 3, h = t & 7;
  int src = ei[e];
  int dst = ei[EDGES + e];
  const float4* qp = (const float4*)(q  + (size_t)dst * HID + h * DH);
  const float4* kp = (const float4*)(kt + (size_t)src * HID + h * DH);
  float s = 0.0f;
#pragma unroll
  for (int i = 0; i < DH / 4; ++i) {
    float4 a = qp[i], b = kp[i];
    s += a.x * b.x + a.y * b.y + a.z * b.z + a.w * b.w;
  }
  s *= prel[h] * 0.17677669529663687f;   // 1/sqrt(32)
  score[t] = s;
  atomicMaxF32(mx + (size_t)dst * HEADS + h, s);
}

__global__ void edge_exp_kernel(const int* __restrict__ ei,
                                float* __restrict__ score,
                                const float* __restrict__ mx,
                                float* __restrict__ den)
{
  int t = blockIdx.x * blockDim.x + threadIdx.x;
  if (t >= EDGES * HEADS) return;
  int e = t >> 3, h = t & 7;
  int dst = ei[EDGES + e];
  float ex = __expf(score[t] - mx[(size_t)dst * HEADS + h]);
  score[t] = ex;
  atomicAdd(den + (size_t)dst * HEADS + h, ex);
}

__global__ void edge_agg_kernel(const int* __restrict__ ei,
                                const float* __restrict__ score,
                                const float* __restrict__ den,
                                const float* __restrict__ vt,
                                float* __restrict__ agg)
{
  int t = blockIdx.x * blockDim.x + threadIdx.x;
  if (t >= EDGES * HEADS) return;
  int e = t >> 3, h = t & 7;
  int src = ei[e];
  int dst = ei[EDGES + e];
  float alpha = score[t] / (den[(size_t)dst * HEADS + h] + 1e-16f);
  const float* vp = vt  + (size_t)src * HID + h * DH;
  float*       ap = agg + (size_t)dst * HID + h * DH;
#pragma unroll
  for (int d = 0; d < DH; ++d) atomicAdd(ap + d, vp[d] * alpha);
}

// ---------------------------------------------------------------------------
// Orchestration
// ---------------------------------------------------------------------------
extern "C" void kernel_launch(void* const* d_in, const int* in_sizes, int n_in,
                              void* d_out, int out_size, void* d_ws, size_t ws_size,
                              hipStream_t stream) {
  const float* x_flow = (const float*)d_in[0];
  const float* x_host = (const float*)d_in[1];
  const int*   ei[3]  = {(const int*)d_in[2], (const int*)d_in[3], (const int*)d_in[4]};
  const float* lin0_W = (const float*)d_in[5];
  const float* lin0_b = (const float*)d_in[6];
  const float* kb = (const float*)d_in[8];
  const float* qb = (const float*)d_in[10];
  const float* vb = (const float*)d_in[12];
  const float* ab = (const float*)d_in[14];
  const float* skip  = (const float*)d_in[15];
  const float* p_rel = (const float*)d_in[18];
  const float* lob   = (const float*)d_in[20];
  float* out = (float*)d_out;
  (void)ws_size; (void)in_sizes; (void)n_in; (void)out_size;

  char* wsb = (char*)d_ws;
  size_t off = 0;
  auto carveF = [&](size_t elems) {
    float* p = (float*)(wsb + off);
    off += ((elems * 4 + 255) & ~(size_t)255);
    return p;
  };
  auto carveB = [&](size_t elems) {
    __bf16* p = (__bf16*)(wsb + off);
    off += ((elems * 2 + 255) & ~(size_t)255);
    return p;
  };

  // f32 state
  float* xs[2]  = {carveF((size_t)NF_N * HID), carveF((size_t)NH_N * HID)};
  float* qx[2]  = {carveF((size_t)NF_N * HID), carveF((size_t)NH_N * HID)};
  float* agg[2] = {carveF((size_t)NF_N * HID), carveF((size_t)NH_N * HID)};
  float* kt    = carveF((size_t)NF_N * HID);
  float* vt    = carveF((size_t)NF_N * HID);
  float* score = carveF((size_t)EDGES * HEADS);
  float* mx    = carveF((size_t)NF_N * HEADS);
  float* den   = carveF((size_t)NF_N * HEADS);
  // bf16 mirrors / converted operands
  __bf16* xinbf[2]  = {carveB((size_t)NF_N * IN_DIM), carveB((size_t)NH_N * IN_DIM)};
  __bf16* xbf[2]    = {carveB((size_t)NF_N * HID), carveB((size_t)NH_N * HID)};
  __bf16* kxbf[2]   = {carveB((size_t)NF_N * HID), carveB((size_t)NH_N * HID)};
  __bf16* vxbf[2]   = {carveB((size_t)NF_N * HID), carveB((size_t)NH_N * HID)};
  __bf16* aggbf[2]  = {carveB((size_t)NF_N * HID), carveB((size_t)NH_N * HID)};
  __bf16* lin0Wbf = carveB((size_t)2 * IN_DIM * HID);
  __bf16* kWbf  = carveB((size_t)LAYERS * 2 * HID * HID);
  __bf16* qWbf  = carveB((size_t)LAYERS * 2 * HID * HID);
  __bf16* vWbf  = carveB((size_t)LAYERS * 2 * HID * HID);
  __bf16* aWbf  = carveB((size_t)LAYERS * 2 * HID * HID);
  __bf16* aRelbf = carveB((size_t)LAYERS * 3 * HEADS * DH * DH);
  __bf16* mRelbf = carveB((size_t)LAYERS * 3 * HEADS * DH * DH);
  __bf16* loWbf  = carveB((size_t)HID * OUT_DIM);

  auto gemm = [&](const __bf16* A, const __bf16* W, const float* bias,
                  const float* Xold, const float* skipP, int skipIdx,
                  float* C, __bf16* Cbf,
                  int M, int K, int N, int lda, int ldw, int ldc,
                  int aOffZ, int wOffZ, int cOffZ, int Z, int epi) {
    dim3 grid((N + BN - 1) / BN, (M + BM - 1) / BM, Z);
    wmma_gemm_kernel<<<grid, 256, 0, stream>>>(A, W, bias, Xold, skipP, skipIdx,
        C, Cbf, M, K, N, lda, ldw, ldc, aOffZ, wOffZ, cOffZ, epi);
  };
  auto fill = [&](float* p, size_t n, float v) {
    fill_kernel<<<(unsigned)((n + 255) / 256), 256, 0, stream>>>(p, (int)n, v);
  };
  auto cvt = [&](const float* s, __bf16* d, size_t n, int act) {
    convert_bf16_kernel<<<(unsigned)((n + 255) / 256), 256, 0, stream>>>(s, d, (int)n, act);
  };

  const int Ns[2] = {NF_N, NH_N};
  const int relSrc[3] = {0, 1, 0}, relDst[3] = {1, 0, 0};
  const int eBlocks = (EDGES * HEADS + 255) / 256;

  // ---- one-time (per call) bf16 conversions ----
  cvt(x_flow, xinbf[0], (size_t)NF_N * IN_DIM, 0);
  cvt(x_host, xinbf[1], (size_t)NH_N * IN_DIM, 0);
  cvt(lin0_W, lin0Wbf, (size_t)2 * IN_DIM * HID, 0);
  cvt((const float*)d_in[7],  kWbf, (size_t)LAYERS * 2 * HID * HID, 0);
  cvt((const float*)d_in[9],  qWbf, (size_t)LAYERS * 2 * HID * HID, 0);
  cvt((const float*)d_in[11], vWbf, (size_t)LAYERS * 2 * HID * HID, 0);
  cvt((const float*)d_in[13], aWbf, (size_t)LAYERS * 2 * HID * HID, 0);
  cvt((const float*)d_in[16], aRelbf, (size_t)LAYERS * 3 * HEADS * DH * DH, 0);
  cvt((const float*)d_in[17], mRelbf, (size_t)LAYERS * 3 * HEADS * DH * DH, 0);
  cvt((const float*)d_in[19], loWbf, (size_t)HID * OUT_DIM, 0);

  // ---- input projection + relu (f32 state + bf16 mirror) ----
  gemm(xinbf[0], lin0Wbf,                lin0_b,       nullptr, nullptr, 0,
       xs[0], xbf[0], NF_N, IN_DIM, HID, IN_DIM, HID, HID, 0, 0, 0, 1, EPI_RELU);
  gemm(xinbf[1], lin0Wbf + IN_DIM * HID, lin0_b + HID, nullptr, nullptr, 0,
       xs[1], xbf[1], NH_N, IN_DIM, HID, IN_DIM, HID, HID, 0, 0, 0, 1, EPI_RELU);

  for (int l = 0; l < LAYERS; ++l) {
    for (int t = 0; t < 2; ++t) {
      size_t wOff = (size_t)(l * 2 + t) * HID * HID;
      size_t bOff = (size_t)(l * 2 + t) * HID;
      // k, v: only bf16 needed (consumed by relation transforms); q: f32 (edge phase)
      gemm(xbf[t], kWbf + wOff, kb + bOff, nullptr, nullptr, 0, nullptr, kxbf[t],
           Ns[t], HID, HID, HID, HID, HID, 0, 0, 0, 1, EPI_NONE);
      gemm(xbf[t], qWbf + wOff, qb + bOff, nullptr, nullptr, 0, qx[t], nullptr,
           Ns[t], HID, HID, HID, HID, HID, 0, 0, 0, 1, EPI_NONE);
      gemm(xbf[t], vWbf + wOff, vb + bOff, nullptr, nullptr, 0, nullptr, vxbf[t],
           Ns[t], HID, HID, HID, HID, HID, 0, 0, 0, 1, EPI_NONE);
    }
    fill(agg[0], (size_t)NF_N * HID, 0.0f);
    fill(agg[1], (size_t)NH_N * HID, 0.0f);

    for (int r = 0; r < 3; ++r) {
      int s = relSrc[r], d = relDst[r];
      const __bf16* Ar = aRelbf + (size_t)(l * 3 + r) * HEADS * DH * DH;
      const __bf16* Mr = mRelbf + (size_t)(l * 3 + r) * HEADS * DH * DH;
      const float*  Pr = p_rel  + (size_t)(l * 3 + r) * HEADS;

      // per-head 32x32 relation transforms (blockIdx.z = head)
      gemm(kxbf[s], Ar, nullptr, nullptr, nullptr, 0, kt, nullptr,
           Ns[s], DH, DH, HID, DH, HID, DH, DH * DH, DH, HEADS, EPI_NONE);
      gemm(vxbf[s], Mr, nullptr, nullptr, nullptr, 0, vt, nullptr,
           Ns[s], DH, DH, HID, DH, HID, DH, DH * DH, DH, HEADS, EPI_NONE);

      fill(mx,  (size_t)Ns[d] * HEADS, -3.0e38f);
      fill(den, (size_t)Ns[d] * HEADS, 0.0f);

      edge_score_kernel<<<eBlocks, 256, 0, stream>>>(ei[r], qx[d], kt, Pr, score, mx);
      edge_exp_kernel  <<<eBlocks, 256, 0, stream>>>(ei[r], score, mx, den);
      edge_agg_kernel  <<<eBlocks, 256, 0, stream>>>(ei[r], score, den, vt, agg[d]);
    }

    // gelu(agg) -> bf16, then @ aW + ab with sigmoid-skip blend (in place)
    for (int t = 0; t < 2; ++t) {
      size_t wOff = (size_t)(l * 2 + t) * HID * HID;
      size_t bOff = (size_t)(l * 2 + t) * HID;
      cvt(agg[t], aggbf[t], (size_t)Ns[t] * HID, 1);
      gemm(aggbf[t], aWbf + wOff, ab + bOff, xs[t], skip, l * 2 + t,
           xs[t], xbf[t], Ns[t], HID, HID, HID, HID, HID, 0, 0, 0, 1, EPI_BLEND);
    }
  }

  // final projection of flow nodes
  gemm(xbf[0], loWbf, lob, nullptr, nullptr, 0, out, nullptr,
       NF_N, HID, OUT_DIM, HID, OUT_DIM, OUT_DIM, 0, 0, 0, 1, EPI_NONE);
}
